// MoELayer_11330123727002
// MI455X (gfx1250) — matrix operand order repaired and checked by reading, not compiled, not run on hardware
//
#include <hip/hip_runtime.h>
#include <hip/hip_bf16.h>
#include <math.h>

// ---------------------------------------------------------------------------
// MoE (top-2 of 8 experts), D=1024, F=4096, routed bf16-WMMA implementation.
// ---------------------------------------------------------------------------

typedef __attribute__((ext_vector_type(16))) __bf16      v16bf;
typedef __attribute__((ext_vector_type(8)))  float       v8f;
typedef __attribute__((ext_vector_type(4)))  unsigned int v4u;

#define D_MODEL   1024
#define D_FF      4096
#define N_EXP     8
#define TILE_M    128
#define TILE_N    128
#define TILE_K    32
#define LDS_STR   48          // bf16 elements per LDS row (16B aligned, padded)

__device__ __forceinline__ unsigned pack_bf16(float a, float b)
{
    const unsigned short lo = __builtin_bit_cast(unsigned short, (__bf16)a);
    const unsigned short hi = __builtin_bit_cast(unsigned short, (__bf16)b);
    return ((unsigned)hi << 16) | (unsigned)lo;
}

// ========================= Kernel 1: router ================================
// One wave (32 lanes) per token: 8 logits, softmax, top-2, gate, bucket append.
__global__ void __launch_bounds__(256)
moe_router(const float* __restrict__ x, const float* __restrict__ Wr,
           const float* __restrict__ br, int* __restrict__ counts,
           int* __restrict__ bucket, float* __restrict__ gates, int T)
{
    const int wave = threadIdx.x >> 5;
    const int lane = threadIdx.x & 31;
    const int t = blockIdx.x * 8 + wave;
    if (t >= T) return;

    float acc[N_EXP];
#pragma unroll
    for (int e = 0; e < N_EXP; ++e) acc[e] = 0.f;

    const float* xrow = x + (size_t)t * D_MODEL;
    for (int d = lane; d < D_MODEL; d += 32) {
        const float xv = xrow[d];
        const float4* wr = (const float4*)(Wr + (size_t)d * N_EXP);
        const float4 w0 = wr[0], w1 = wr[1];
        acc[0] += xv * w0.x; acc[1] += xv * w0.y;
        acc[2] += xv * w0.z; acc[3] += xv * w0.w;
        acc[4] += xv * w1.x; acc[5] += xv * w1.y;
        acc[6] += xv * w1.z; acc[7] += xv * w1.w;
    }
#pragma unroll
    for (int e = 0; e < N_EXP; ++e)
#pragma unroll
        for (int off = 16; off > 0; off >>= 1)
            acc[e] += __shfl_xor(acc[e], off, 32);

    if (lane == 0) {
        float lg[N_EXP], mx = -1e30f;
#pragma unroll
        for (int e = 0; e < N_EXP; ++e) { lg[e] = acc[e] + br[e]; mx = fmaxf(mx, lg[e]); }
#pragma unroll
        for (int e = 0; e < N_EXP; ++e) lg[e] = __expf(lg[e] - mx);
        // softmax normalizer cancels in top-2 gate renormalization
        int e1 = 0; float v1 = lg[0];
#pragma unroll
        for (int e = 1; e < N_EXP; ++e) if (lg[e] > v1) { v1 = lg[e]; e1 = e; }
        int e2 = (e1 == 0) ? 1 : 0; float v2 = lg[e2];
#pragma unroll
        for (int e = 0; e < N_EXP; ++e)
            if (e != e1 && lg[e] > v2) { v2 = lg[e]; e2 = e; }
        const float inv = 1.f / (v1 + v2);
        const int p1 = atomicAdd(&counts[e1], 1);
        bucket[e1 * T + p1] = t; gates[e1 * T + p1] = v1 * inv;
        const int p2 = atomicAdd(&counts[e2], 1);
        bucket[e2 * T + p2] = t; gates[e2 * T + p2] = v2 * inv;
    }
}

// ========================= Kernel 2: prefix ================================
__global__ void moe_prefix(const int* __restrict__ counts, int* __restrict__ offs)
{
    if (threadIdx.x == 0) {
        int s = 0;
        for (int e = 0; e < N_EXP; ++e) { offs[e] = s; s += counts[e]; }
        offs[N_EXP] = s;
    }
}

// ===================== WMMA fragment helpers ===============================
__device__ __forceinline__ v16bf frag_a(const __bf16* rowp, int hA)
{
    // A 16x32 bf16: lanes 0-15 hold K{0-7,16-23}, lanes 16-31 hold K{8-15,24-31}
    union { v16bf v; v4u q[2]; } f;
    const v4u* p = (const v4u*)rowp;
    f.q[0] = p[hA];        // K = hA*8 .. +8
    f.q[1] = p[2 + hA];    // K = 16 + hA*8 .. +8
    return f.v;
}
__device__ __forceinline__ v16bf frag_b(const __bf16* rowp, int hA)
{
    // B 32x16 bf16: lane = N, lanes 0-15 K{0-7,16-23}, lanes 16-31 K{8-15,24-31}
    union { v16bf v; v4u q[2]; } f;
    const v4u* p = (const v4u*)rowp;
    f.q[0] = p[hA];
    f.q[1] = p[2 + hA];
    return f.v;
}

// ================= Kernel 3: H = gelu(X @ W1 + b1), bf16 out ===============
__global__ void __launch_bounds__(256)
moe_ffn1(const float* __restrict__ x, const float* __restrict__ W1,
         const float* __restrict__ b1, const int* __restrict__ counts,
         const int* __restrict__ offs, const int* __restrict__ bucket,
         __bf16* __restrict__ H, int T, int tilesPerExp)
{
    const int e    = blockIdx.x / tilesPerExp;
    const int tile = blockIdx.x % tilesPerExp;
    const int cnt  = counts[e];
    const int tbase = tile * TILE_M;
    if (tbase >= cnt) return;                    // uniform early exit
    const int nbase = blockIdx.y * TILE_N;
    const int hbase = offs[e] + tbase;

    __shared__ __bf16 Xs[TILE_M][LDS_STR];
    __shared__ __bf16 Ws[TILE_N][LDS_STR];
    __shared__ int    rowTok[TILE_M];

    if (threadIdx.x < TILE_M) {
        int pos = tbase + (int)threadIdx.x;
        if (pos >= cnt) pos = cnt - 1;           // clamp: always a real token id
        rowTok[threadIdx.x] = bucket[e * T + pos];
    }

    const int lane = threadIdx.x & 31;
    const int wv   = threadIdx.x >> 5;
    const int mw   = (wv >> 1) * 32;             // 4 waves along M (32 rows each)
    const int nw   = (wv & 1) * 64;              // 2 waves along N (64 cols each)
    const int rA   = lane & 15;
    const int hA   = lane >> 4;

    v8f acc[2][4];
#pragma unroll
    for (int mi = 0; mi < 2; ++mi)
#pragma unroll
        for (int ni = 0; ni < 4; ++ni)
#pragma unroll
            for (int r = 0; r < 8; ++r) acc[mi][ni][r] = 0.f;

    const float* wbase = W1 + (size_t)e * D_MODEL * D_FF + nbase;

    for (int kb = 0; kb < D_MODEL; kb += TILE_K) {
        __syncthreads();
        {   // stage gathered X tile (f32 -> bf16, packed dword stores)
            const int r  = threadIdx.x >> 1;
            const int ks = (threadIdx.x & 1) * 16;
            const float4* src = (const float4*)(x + (size_t)rowTok[r] * D_MODEL + kb + ks);
#pragma unroll
            for (int j = 0; j < 4; ++j) {
                const float4 q = src[j];
                *(unsigned*)&Xs[r][ks + 4 * j]     = pack_bf16(q.x, q.y);
                *(unsigned*)&Xs[r][ks + 4 * j + 2] = pack_bf16(q.z, q.w);
            }
        }
        {   // stage W1 chunk transposed: Ws[n][k], packed k-pairs per dword
            const float* wsrc = wbase + (size_t)kb * D_FF;
#pragma unroll
            for (int i = 0; i < 8; ++i) {
                const int idx = threadIdx.x + i * 256;       // 0..2047 (n, k-pair)
                const int n  = idx & 127;
                const int k2 = (idx >> 7) * 2;
                const float f0 = wsrc[(size_t)k2 * D_FF + n];
                const float f1 = wsrc[(size_t)(k2 + 1) * D_FF + n];
                *(unsigned*)&Ws[n][k2] = pack_bf16(f0, f1);
            }
            if (kb + TILE_K < D_MODEL)   // global_prefetch_b8 next weight tile
                __builtin_prefetch(wsrc + (size_t)TILE_K * D_FF + (threadIdx.x & 127), 0, 1);
        }
        __syncthreads();

        v16bf a0 = frag_a(&Xs[mw + rA][0], hA);
        v16bf a1 = frag_a(&Xs[mw + 16 + rA][0], hA);
#pragma unroll
        for (int ni = 0; ni < 4; ++ni) {
            v16bf b = frag_b(&Ws[nw + ni * 16 + rA][0], hA);
            acc[0][ni] = __builtin_amdgcn_wmma_f32_16x16x32_bf16(
                false, a0, false, b, (short)0, acc[0][ni], false, false);
            acc[1][ni] = __builtin_amdgcn_wmma_f32_16x16x32_bf16(
                false, a1, false, b, (short)0, acc[1][ni], false, false);
        }
    }

    // epilogue: bias + exact GELU, store bf16 H (skip invalid rows)
    const float* b1p = b1 + (size_t)e * D_FF + nbase;
#pragma unroll
    for (int mi = 0; mi < 2; ++mi)
#pragma unroll
        for (int ni = 0; ni < 4; ++ni) {
            const int n = nw + ni * 16 + rA;
            const float bias = b1p[n];
#pragma unroll
            for (int r = 0; r < 8; ++r) {
                const int m = mw + mi * 16 + hA * 8 + r;
                float v = acc[mi][ni][r] + bias;
                v = 0.5f * v * (1.f + erff(v * 0.70710678118654752f));
                if (tbase + m < cnt)
                    H[(size_t)(hbase + m) * D_FF + nbase + n] = (__bf16)v;
            }
        }
}

// ============ Kernel 4: out += gate * (H @ W2 + b2) (atomic scatter) =======
__global__ void __launch_bounds__(256)
moe_ffn2(const __bf16* __restrict__ H, const float* __restrict__ W2,
         const float* __restrict__ b2, const int* __restrict__ counts,
         const int* __restrict__ offs, const int* __restrict__ bucket,
         const float* __restrict__ gates, float* __restrict__ out,
         int T, int tilesPerExp)
{
    const int e    = blockIdx.x / tilesPerExp;
    const int tile = blockIdx.x % tilesPerExp;
    const int cnt  = counts[e];
    const int tbase = tile * TILE_M;
    if (tbase >= cnt) return;
    const int nbase = blockIdx.y * TILE_N;
    const int hbase = offs[e] + tbase;

    __shared__ __bf16 Xs[TILE_M][LDS_STR];
    __shared__ __bf16 Ws[TILE_N][LDS_STR];
    __shared__ int    rowTok[TILE_M];
    __shared__ float  rowGate[TILE_M];

    if (threadIdx.x < TILE_M) {
        int pos = tbase + (int)threadIdx.x;
        if (pos >= cnt) pos = cnt - 1;
        rowTok[threadIdx.x]  = bucket[e * T + pos];
        rowGate[threadIdx.x] = gates[e * T + pos];
    }

    const int lane = threadIdx.x & 31;
    const int wv   = threadIdx.x >> 5;
    const int mw   = (wv >> 1) * 32;
    const int nw   = (wv & 1) * 64;
    const int rA   = lane & 15;
    const int hA   = lane >> 4;

    v8f acc[2][4];
#pragma unroll
    for (int mi = 0; mi < 2; ++mi)
#pragma unroll
        for (int ni = 0; ni < 4; ++ni)
#pragma unroll
            for (int r = 0; r < 8; ++r) acc[mi][ni][r] = 0.f;

    const float* wbase = W2 + (size_t)e * D_FF * D_MODEL + nbase;

    for (int kb = 0; kb < D_FF; kb += TILE_K) {
        __syncthreads();   // all waves done reading previous tile (dscnt==0 there)
        {   // stage H tile: pure bf16 copy -> async DMA to LDS, no VGPR round-trip
            const int r  = threadIdx.x >> 1;
            const int ks = (threadIdx.x & 1) * 16;
            const __bf16* gsrc = H + (size_t)(hbase + r) * D_FF + kb + ks;
            // low 32 bits of the flat pointer = wave-relative LDS byte offset
            const unsigned lds0 = (unsigned)(size_t)&Xs[r][ks];
            // INST_OFFSET is added to BOTH lds and global addresses (ISA 15.18.3)
            asm volatile(
                "global_load_async_to_lds_b128 %0, %1, off\n\t"
                "global_load_async_to_lds_b128 %0, %1, off offset:16"
                :: "v"(lds0), "v"((unsigned long long)(size_t)gsrc)
                : "memory");
        }
        {   // stage W2 chunk transposed (f32 -> bf16, packed k-pairs) on VALU,
            // overlapping with the async H DMA above
            const float* wsrc = wbase + (size_t)kb * D_MODEL;
#pragma unroll
            for (int i = 0; i < 8; ++i) {
                const int idx = threadIdx.x + i * 256;
                const int n  = idx & 127;
                const int k2 = (idx >> 7) * 2;
                const float f0 = wsrc[(size_t)k2 * D_MODEL + n];
                const float f1 = wsrc[(size_t)(k2 + 1) * D_MODEL + n];
                *(unsigned*)&Ws[n][k2] = pack_bf16(f0, f1);
            }
            if (kb + TILE_K < D_FF)
                __builtin_prefetch(wsrc + (size_t)TILE_K * D_MODEL + (threadIdx.x & 127), 0, 1);
        }
        asm volatile("s_wait_asynccnt 0x0" ::: "memory");  // our async writes landed
        __syncthreads();                                   // everyone's writes visible

        v16bf a0 = frag_a(&Xs[mw + rA][0], hA);
        v16bf a1 = frag_a(&Xs[mw + 16 + rA][0], hA);
#pragma unroll
        for (int ni = 0; ni < 4; ++ni) {
            v16bf b = frag_b(&Ws[nw + ni * 16 + rA][0], hA);
            acc[0][ni] = __builtin_amdgcn_wmma_f32_16x16x32_bf16(
                false, a0, false, b, (short)0, acc[0][ni], false, false);
            acc[1][ni] = __builtin_amdgcn_wmma_f32_16x16x32_bf16(
                false, a1, false, b, (short)0, acc[1][ni], false, false);
        }
    }

    const float* b2p = b2 + (size_t)e * D_MODEL + nbase;
#pragma unroll
    for (int mi = 0; mi < 2; ++mi)
#pragma unroll
        for (int ni = 0; ni < 4; ++ni) {
            const int n = nw + ni * 16 + rA;
            const float bias = b2p[n];
#pragma unroll
            for (int r = 0; r < 8; ++r) {
                const int m = mw + mi * 16 + hA * 8 + r;
                if (tbase + m < cnt) {
                    const float v = (acc[mi][ni][r] + bias) * rowGate[m];
                    atomicAdd(&out[(size_t)rowTok[m] * D_MODEL + nbase + n], v);
                }
            }
        }
}

// =============================== launcher ==================================
extern "C" void kernel_launch(void* const* d_in, const int* in_sizes, int n_in,
                              void* d_out, int out_size, void* d_ws, size_t ws_size,
                              hipStream_t stream)
{
    const float* x  = (const float*)d_in[0];
    const float* Wr = (const float*)d_in[1];
    const float* br = (const float*)d_in[2];
    const float* W1 = (const float*)d_in[3];
    const float* b1 = (const float*)d_in[4];
    const float* W2 = (const float*)d_in[5];
    const float* b2 = (const float*)d_in[6];
    float* out = (float*)d_out;

    const int T = in_sizes[0] / D_MODEL;              // 4096 tokens

    // workspace layout
    char* ws = (char*)d_ws;
    int*   counts = (int*)ws;                          // 8 ints
    int*   offs   = (int*)(ws + 64);                   // 9 ints
    int*   bucket = (int*)(ws + 256);                  // E*T ints
    float* gates  = (float*)(ws + 256 + (size_t)N_EXP * T * sizeof(int));
    size_t hoff = 256 + (size_t)N_EXP * T * (sizeof(int) + sizeof(float));
    hoff = (hoff + 255) & ~(size_t)255;
    __bf16* H = (__bf16*)(ws + hoff);                  // (2T + 128 slack) x D_FF bf16

    hipMemsetAsync(counts, 0, 64, stream);
    hipMemsetAsync(out, 0, (size_t)out_size * sizeof(float), stream);

    moe_router<<<dim3((T + 7) / 8), dim3(256), 0, stream>>>(
        x, Wr, br, counts, bucket, gates, T);
    moe_prefix<<<dim3(1), dim3(32), 0, stream>>>(counts, offs);

    const int tilesPerExp = (T + TILE_M - 1) / TILE_M;
    moe_ffn1<<<dim3(N_EXP * tilesPerExp, D_FF / TILE_N), dim3(256), 0, stream>>>(
        x, W1, b1, counts, offs, bucket, H, T, tilesPerExp);
    moe_ffn2<<<dim3(N_EXP * tilesPerExp, D_MODEL / TILE_N), dim3(256), 0, stream>>>(
        H, W2, b2, counts, offs, bucket, gates, out, T, tilesPerExp);
}
